// DualMIMamba_CrossSSM_69028714381788
// MI455X (gfx1250) — matrix operand order repaired and checked by reading, not compiled, not run on hardware
//
#include <hip/hip_runtime.h>
#include <math.h>

// ---------------------------------------------------------------------------
// CDNA5 (gfx1250) WMMA types
// ---------------------------------------------------------------------------
typedef __attribute__((ext_vector_type(16))) _Float16 v16h;
typedef __attribute__((ext_vector_type(8)))  float    v8f;

// Async global->LDS copy (CDNA5). Fall back to load+ds_store if the builtin
// is not available on this toolchain; guarded so host pass never sees it.
#if defined(__gfx1250__)
#if __has_builtin(__builtin_amdgcn_global_load_async_to_lds_b32) && \
    __has_builtin(__builtin_amdgcn_s_wait_asynccnt)
#define ASYNC_LDS 1
#endif
#endif

#if defined(ASYNC_LDS)
typedef __attribute__((address_space(1))) int* glb_ip;
typedef __attribute__((address_space(3))) int* lds_ip;
#endif

__device__ __forceinline__ void copy_g2l_f32(const float* g, float* l) {
#if defined(ASYNC_LDS)
  __builtin_amdgcn_global_load_async_to_lds_b32(
      (glb_ip)(unsigned long long)g, (lds_ip)l, 0, 0);
#else
  *l = *g;
#endif
}
// wait: tile in flight behind the current one (20 async ops/thread/tile)
__device__ __forceinline__ void copy_join_prev() {
#if defined(ASYNC_LDS)
  __builtin_amdgcn_s_wait_asynccnt(20);
#endif
}
__device__ __forceinline__ void copy_join_all() {
#if defined(ASYNC_LDS)
  __builtin_amdgcn_s_wait_asynccnt(0);
#endif
}

// ---------------------------------------------------------------------------
// Parameter flattening indices (top level insertion order, nested pytree
// order: dict keys sorted, lists in order). 0: x_a2c  1: x_a4c  then params.
// ---------------------------------------------------------------------------
enum {
  IN_XA4C = 1,
  CLS_BN1_B = 2, CLS_BN1_G, CLS_BN2_B, CLS_BN2_G, CLS_FC1_B, CLS_FC1_W,
  CLS_FC2_B, CLS_FC2_W, CLS_FC3_B, CLS_FC3_W,                       // 2..11
  ENC_DOWN_BASE = 12,                                               // 3 x {b,w}
  ENC_GSC_BASE  = 18,                                               // 4 x {b1..b4,w1..w4}
  ENC_MLP3_B1 = 50, ENC_MLP3_B2, ENC_MLP3_W1, ENC_MLP3_W2,          // 50..53
  ENC_STAGE_BASE = 54,                                              // 8 layers x 25
  ENC_STEM_B = 254, ENC_STEM_W = 255,
  HEAD_C1_B = 256, HEAD_C1_W, HEAD_C2_B, HEAD_C2_W, HEAD_RES_B, HEAD_RES_W
};
// per mamba layer (25): 0 ln_b, 1 ln_g, 2..8 bwd, 9..15 fwd, 16 in_proj_w,
// 17 out_proj_w, 18..24 sl.  Dir(7): A_log, D, conv_b, conv_w, dtproj_b,
// dtproj_w, xproj_w.

// ---------------------------------------------------------------------------
// Strided GEMM on v_wmma_f32_16x16x32_f16 with double-buffered, async-copied,
// fragment-ordered LDS tiles. Block = 128 threads (4 waves).
// Tile: 16(M) x 64(N), K step 32; copy of tile t+1 overlaps WMMA on tile t.
// C[m,n] = act( sum_k A[m,k]*B[k,n] + bias[m] + Cres[m,n] ), per batch z.
// ---------------------------------------------------------------------------
__global__ void k_gemm_wmma(const float* __restrict__ A, const float* __restrict__ Bm,
                            float* __restrict__ C, const float* __restrict__ bias,
                            const float* __restrict__ Cres,
                            int M, int N, int K,
                            long sAm, long sAk, long batchA,
                            long sBk, long sBn, long batchB,
                            long sCm, long sCn, long batchC,
                            int act) {
  __shared__ float sA[2][512];    // [buf][lane(32)][i(16)] fragment-ordered f32
  __shared__ float sB[2][2048];   // [buf][wave(4)][lane(32)][i(16)]
  int tid = threadIdx.x;
  int wave = tid >> 5;
  int lane = tid & 31;
  int m0 = blockIdx.x * 16;
  int n0blk = blockIdx.y * 64;
  long zb = blockIdx.z;
  const float* Ab = A + zb * batchA;
  const float* Bb = Bm + zb * batchB;
  float*       Cb = C + zb * batchC;
  const float* Rb = Cres ? Cres + zb * batchC : nullptr;
  bool hiv = lane >= 16;
  bool fullMN = (m0 + 16 <= M) && (n0blk + 64 <= N);   // block-uniform

  v8f acc = {};

  // Branch-free tile stage: exactly 20 async b32 copies per thread.
  // OOB elements read a safe in-tile address; they are zeroed at build time.
  auto stage = [&](int k0, int buf) {
#pragma unroll
    for (int q = 0; q < 4; ++q) {          // A: 512 elems
      int e = tid + 128 * q;
      int ln = e >> 4, i = e & 15;
      int v = i >> 1, par = i & 1;
      int kk = (v < 4 ? 2 * v : 16 + 2 * (v - 4)) + par + ((ln >= 16) ? 8 : 0);
      int gm = m0 + (ln & 15), gk = k0 + kk;
      bool ok = (gm < M) && (gk < K);
      long off = ok ? ((long)gm * sAm + (long)gk * sAk)
                    : ((long)m0 * sAm + (long)k0 * sAk);
      copy_g2l_f32(Ab + off, &sA[buf][e]);
    }
#pragma unroll
    for (int q = 0; q < 16; ++q) {         // B: 2048 elems
      int e = tid + 128 * q;
      int w = e >> 9, r2 = e & 511, ln = r2 >> 4, i = r2 & 15;
      int gn = n0blk + w * 16 + (ln & 15);
      int gk = k0 + (i < 8 ? i : 8 + i) + ((ln >= 16) ? 8 : 0);
      bool ok = (gn < N) && (gk < K);
      long off = ok ? ((long)gk * sBk + (long)gn * sBn)
                    : ((long)k0 * sBk + (long)n0blk * sBn);
      copy_g2l_f32(Bb + off, &sB[buf][e]);
    }
  };

  int mA = m0 + (lane & 15);
  int nB = n0blk + wave * 16 + (lane & 15);

  auto compute = [&](int k0, int buf) {
    const float* pa = &sA[buf][lane * 16];
    const float* pb = &sB[buf][wave * 512 + lane * 16];
    v16h af, bf;
    if (fullMN && (k0 + 32 <= K)) {        // fast path: no masks (uniform branch)
#pragma unroll
      for (int i = 0; i < 16; ++i) {
        af[i] = (_Float16)pa[i];
        bf[i] = (_Float16)pb[i];
      }
    } else {
#pragma unroll
      for (int i = 0; i < 16; ++i) {
        int v = i >> 1, par = i & 1;
        int kkA = (v < 4 ? 2 * v : 16 + 2 * (v - 4)) + par + (hiv ? 8 : 0);
        int kkB = (i < 8 ? i : 8 + i) + (hiv ? 8 : 0);
        bool okA = (mA < M) && (k0 + kkA < K);
        bool okB = (nB < N) && (k0 + kkB < K);
        af[i] = okA ? (_Float16)pa[i] : (_Float16)0.f;
        bf[i] = okB ? (_Float16)pb[i] : (_Float16)0.f;
      }
    }
    acc = __builtin_amdgcn_wmma_f32_16x16x32_f16(false, af, false, bf,
                                                 (short)0, acc, false, false);
  };

  int nk = (K + 31) / 32;
  stage(0, 0);
  for (int t = 0; t < nk; ++t) {
    if (t + 1 < nk) {
      stage((t + 1) * 32, (t + 1) & 1);    // prefetch next tile into other buffer
      copy_join_prev();                    // tile t landed; t+1 still in flight
    } else {
      copy_join_all();
    }
    __syncthreads();
    compute(t * 32, t & 1);
    __syncthreads();                       // protect buffer before re-stage
  }

  __builtin_amdgcn_s_wait_tensorcnt(0);    // CDNA5 split-counter wait (no-op here)
#pragma unroll
  for (int r = 0; r < 8; ++r) {
    int m = m0 + r + (hiv ? 8 : 0);
    if (m < M && nB < N) {
      float v = acc[r];
      if (bias) v += bias[m];
      if (Rb)   v += Rb[(long)m * sCm + (long)nB * sCn];
      if (act == 1) v = fmaxf(v, 0.f);
      Cb[(long)m * sCm + (long)nB * sCn] = v;
    }
  }
}

// ---------------------------------------------------------------------------
// Direct 3D convolution (stem 7^3 s2, GSC 3^3, down 2^3 s2, head 3^3)
// ---------------------------------------------------------------------------
__global__ void k_conv3d(const float* __restrict__ in, const float* __restrict__ w,
                         const float* __restrict__ bias, float* __restrict__ out,
                         int Bn, int Ci, int Di, int Hi, int Wi,
                         int Co, int Kk, int stride, int pad,
                         int Do, int Ho, int Wo) {
  long total = (long)Bn * Co * Do * Ho * Wo;
  long idx = (long)blockIdx.x * blockDim.x + threadIdx.x;
  if (idx >= total) return;
  int xo = (int)(idx % Wo); long t = idx / Wo;
  int yo = (int)(t % Ho); t /= Ho;
  int zo = (int)(t % Do); t /= Do;
  int co = (int)(t % Co); int b = (int)(t / Co);
  float accv = bias ? bias[co] : 0.f;
  const float* wc = w + (long)co * Ci * Kk * Kk * Kk;
  for (int ci = 0; ci < Ci; ++ci) {
    const float* inc = in + ((long)b * Ci + ci) * Di * Hi * Wi;
    const float* wci = wc + (long)ci * Kk * Kk * Kk;
    for (int kz = 0; kz < Kk; ++kz) {
      int zi = zo * stride - pad + kz;
      if ((unsigned)zi >= (unsigned)Di) continue;
      for (int ky = 0; ky < Kk; ++ky) {
        int yi = yo * stride - pad + ky;
        if ((unsigned)yi >= (unsigned)Hi) continue;
        const float* row = inc + ((long)zi * Hi + yi) * Wi;
        const float* wr = wci + (kz * Kk + ky) * Kk;
        __builtin_prefetch(row + Wi, 0, 1);  // -> global_prefetch_b8
        for (int kx = 0; kx < Kk; ++kx) {
          int xi = xo * stride - pad + kx;
          if ((unsigned)xi >= (unsigned)Wi) continue;
          accv = fmaf(row[xi], wr[kx], accv);
        }
      }
    }
  }
  out[idx] = accv;
}

// ---------------------------------------------------------------------------
// Instance norm per (b,c) plane of L elems. mode: 0 none, 1 relu, 2 lrelu0.01.
// Optional residual added AFTER activation (GSC tail). In-place safe.
// ---------------------------------------------------------------------------
__global__ void k_instnorm(const float* __restrict__ in, float* __restrict__ out,
                           const float* __restrict__ res, long L, int mode) {
  const float* p = in + (long)blockIdx.x * L;
  float* q = out + (long)blockIdx.x * L;
  const float* r = res ? res + (long)blockIdx.x * L : nullptr;
  __shared__ float s1[256], s2[256];
  float s = 0.f, ss = 0.f;
  for (long i = threadIdx.x; i < L; i += blockDim.x) { float v = p[i]; s += v; ss += v * v; }
  s1[threadIdx.x] = s; s2[threadIdx.x] = ss;
  __syncthreads();
  for (int st = 128; st > 0; st >>= 1) {
    if ((int)threadIdx.x < st) { s1[threadIdx.x] += s1[threadIdx.x + st];
                                 s2[threadIdx.x] += s2[threadIdx.x + st]; }
    __syncthreads();
  }
  float mu = s1[0] / (float)L;
  float var = s2[0] / (float)L - mu * mu;
  float rs = rsqrtf(var + 1e-5f);
  for (long i = threadIdx.x; i < L; i += blockDim.x) {
    float v = (p[i] - mu) * rs;
    if (mode == 1) v = fmaxf(v, 0.f);
    else if (mode == 2) v = v > 0.f ? v : 0.01f * v;
    if (r) v += r[i];
    q[i] = v;
  }
}

// ---------------------------------------------------------------------------
// LayerNorm over channels, reading channel-major (B,C,L), writing token-major
// [b*L+l, C]. One wave per token (wave32).
// ---------------------------------------------------------------------------
__global__ void k_layernorm(const float* __restrict__ x, const float* __restrict__ g,
                            const float* __restrict__ bt, float* __restrict__ tok,
                            int C, long L, long T) {
  long wid = (long)blockIdx.x * (blockDim.x >> 5) + (threadIdx.x >> 5);
  int lane = threadIdx.x & 31;
  if (wid >= T) return;
  long b = wid / L, l = wid % L;
  const float* base = x + (long)b * C * L + l;
  float s = 0.f, ss = 0.f;
  for (int c = lane; c < C; c += 32) { float v = base[(long)c * L]; s += v; ss += v * v; }
  for (int m = 16; m > 0; m >>= 1) { s += __shfl_xor(s, m, 32); ss += __shfl_xor(ss, m, 32); }
  float mu = s / (float)C;
  float rs = rsqrtf(ss / (float)C - mu * mu + 1e-5f);
  float* o = tok + wid * C;
  for (int c = lane; c < C; c += 32)
    o[c] = (base[(long)c * L] - mu) * rs * g[c] + bt[c];
}

// ---------------------------------------------------------------------------
// Causal depthwise conv1d (k=4) + SiLU. xz is (B, 2*dI, L); x half = rows [0,dI).
// ---------------------------------------------------------------------------
__global__ void k_dwconv_silu(const float* __restrict__ xz, const float* __restrict__ w,
                              const float* __restrict__ bias, float* __restrict__ out,
                              int Bn, int dI, long L) {
  long total = (long)Bn * dI * L;
  long idx = (long)blockIdx.x * blockDim.x + threadIdx.x;
  if (idx >= total) return;
  long l = idx % L; long t = idx / L;
  int d = (int)(t % dI); int b = (int)(t / dI);
  const float* xr = xz + ((long)b * 2 * dI + d) * L;
  const float* wr = w + (long)d * 4;
  float a = bias[d];
#pragma unroll
  for (int j = 0; j < 4; ++j) { long li = l - 3 + j; if (li >= 0) a = fmaf(xr[li], wr[j], a); }
  out[idx] = a / (1.f + expf(-a));
}

// delta = softplus(dtproj_w @ dt + b); dt rows are xdbl[0:dtr)
__global__ void k_dtproj_softplus(const float* __restrict__ xdbl, const float* __restrict__ w,
                                  const float* __restrict__ bias, float* __restrict__ delta,
                                  int Bn, int dI, int R, int dtr, long L) {
  long total = (long)Bn * dI * L;
  long idx = (long)blockIdx.x * blockDim.x + threadIdx.x;
  if (idx >= total) return;
  long l = idx % L; long t = idx / L;
  int d = (int)(t % dI); int b = (int)(t / dI);
  const float* xb = xdbl + (long)b * R * L + l;
  const float* wd = w + (long)d * dtr;
  float a = bias[d];
  for (int r = 0; r < dtr; ++r) a = fmaf(wd[r], xb[(long)r * L], a);
  delta[idx] = a > 20.f ? a : log1pf(expf(a));
}

// ---------------------------------------------------------------------------
// Selective scan: wave handles 2 channels (lane half = channel, lane&15 = state n).
// h[l] = exp(dt*A)*h[l-1] + dt*B[l]*u[l];  y = (sum_n h*C[l] + u*D) * silu(z).
// ---------------------------------------------------------------------------
__global__ void k_selscan(const float* __restrict__ u, const float* __restrict__ delta,
                          const float* __restrict__ xdbl, const float* __restrict__ xz,
                          const float* __restrict__ A_log, const float* __restrict__ Dp,
                          float* __restrict__ y, int Bn, int dI, int R, int dtr, long L) {
  int lane = threadIdx.x & 31;
  int pairs = dI >> 1;
  int b = blockIdx.x / pairs;
  int pr = blockIdx.x % pairs;
  int n = lane & 15;
  int d = pr * 2 + (lane >> 4);
  float Av = -expf(A_log[(long)d * 16 + n]);
  float Dv = Dp[d];
  const float* ub = u + ((long)b * dI + d) * L;
  const float* db = delta + ((long)b * dI + d) * L;
  const float* Bb = xdbl + (long)b * R * L + (long)(dtr + n) * L;
  const float* Cb = xdbl + (long)b * R * L + (long)(dtr + 16 + n) * L;
  const float* zb = xz + ((long)b * 2 * dI + dI + d) * L;
  float* yb = y + ((long)b * dI + d) * L;
  float h = 0.f;
  for (long l = 0; l < L; ++l) {
    float dv = db[l], uv = ub[l];
    float a = expf(dv * Av);
    h = fmaf(a, h, dv * Bb[l] * uv);
    float p = h * Cb[l];
    p += __shfl_xor(p, 1, 32);
    p += __shfl_xor(p, 2, 32);
    p += __shfl_xor(p, 4, 32);
    p += __shfl_xor(p, 8, 32);
    if (n == 0) {
      float zv = zb[l];
      yb[l] = (p + uv * Dv) * (zv / (1.f + expf(-zv)));
    }
  }
}

// dst[.., j] = src[.., map(j)]  mode 1: reverse; mode 2: slice gather (j=ln*ns+s <- s*Ln+ln)
__global__ void k_gather(const float* __restrict__ src, float* __restrict__ dst,
                         long rows, long L, int mode, int ns) {
  long total = rows * L;
  long idx = (long)blockIdx.x * blockDim.x + threadIdx.x;
  if (idx >= total) return;
  long j = idx % L; long base = idx - j;
  long sidx;
  if (mode == 1) sidx = L - 1 - j;
  else { long Ln = L / ns; long s = j % ns; long ln = j / ns; sidx = s * Ln + ln; }
  dst[idx] = src[base + sidx];
}

// mode 0: sum = y; 1: sum += y[rev]; 2: sum[j=s*Ln+ln] += y[ln*ns+s]
__global__ void k_accum(const float* __restrict__ y, float* __restrict__ sum,
                        long rows, long L, int mode, int ns) {
  long total = rows * L;
  long idx = (long)blockIdx.x * blockDim.x + threadIdx.x;
  if (idx >= total) return;
  long j = idx % L; long base = idx - j;
  if (mode == 0) sum[idx] = y[idx];
  else if (mode == 1) sum[idx] += y[base + (L - 1 - j)];
  else { long Ln = L / ns; long s = j / Ln; long ln = j % Ln; sum[idx] += y[base + ln * ns + s]; }
}

__global__ void k_add(const float* a, const float* b, float* c, long n) {
  long i = (long)blockIdx.x * blockDim.x + threadIdx.x;
  if (i < n) c[i] = a[i] + b[i];
}
__global__ void k_add_lrelu(const float* a, const float* b, float* c, long n) {
  long i = (long)blockIdx.x * blockDim.x + threadIdx.x;
  if (i < n) { float v = a[i] + b[i]; c[i] = v > 0.f ? v : 0.01f * v; }
}
__global__ void k_gelu(float* x, long n) {
  long i = (long)blockIdx.x * blockDim.x + threadIdx.x;
  if (i < n) {
    float v = x[i];
    x[i] = 0.5f * v * (1.f + tanhf(0.7978845608f * (v + 0.044715f * v * v * v)));
  }
}
__global__ void k_gap(const float* x, float* g, int L, long n) {
  long i = (long)blockIdx.x * blockDim.x + threadIdx.x;
  if (i >= n) return;
  const float* p = x + i * L;
  float s = 0.f;
  for (int j = 0; j < L; ++j) s += p[j];
  g[i] = s / (float)L;
}
// batchnorm over batch of 2, per feature, optional relu; layout [b*F + f]
__global__ void k_bnrelu(float* x, const float* g, const float* b, int F, int relu) {
  int f = blockIdx.x * blockDim.x + threadIdx.x;
  if (f >= F) return;
  float a = x[f], c = x[F + f];
  float mu = 0.5f * (a + c);
  float var = 0.5f * ((a - mu) * (a - mu) + (c - mu) * (c - mu));
  float rs = rsqrtf(var + 1e-5f);
  float ya = (a - mu) * rs * g[f] + b[f];
  float yc = (c - mu) * rs * g[f] + b[f];
  if (relu) { ya = fmaxf(ya, 0.f); yc = fmaxf(yc, 0.f); }
  x[f] = ya; x[F + f] = yc;
}

// ---------------------------------------------------------------------------
// Host orchestration
// ---------------------------------------------------------------------------
extern "C" void kernel_launch(void* const* d_in, const int* in_sizes, int n_in,
                              void* d_out, int out_size, void* d_ws, size_t ws_size,
                              hipStream_t stream) {
  (void)in_sizes; (void)n_in; (void)out_size; (void)ws_size;
  auto P = [&](int i) { return (const float*)d_in[i]; };
  const int Bn = 2;
  static const int FEAT[4] = {48, 96, 192, 384};
  static const int NS[4] = {32, 16, 8, 4};
  static const int DD[4] = {16, 8, 4, 2}, HH[4] = {16, 8, 4, 2}, WW[4] = {32, 16, 8, 4};

  float* ws = (float*)d_ws;
  size_t off = 0;
  auto alloc = [&](size_t n) { float* p = ws + off; off += (n + 63) & ~(size_t)63; return p; };
  const size_t S0 = (size_t)Bn * 48 * 8192;
  float* xA = alloc(S0); float* xB = alloc(S0);
  float* t1 = alloc(S0); float* t2 = alloc(S0); float* t3 = alloc(S0);
  float* tok = alloc(S0);
  const size_t SZ = (size_t)Bn * 192 * 8192;
  float* Z0 = alloc(SZ); float* Z1 = alloc(SZ);
  const size_t SM = (size_t)Bn * 96 * 8192;
  float* Mu = alloc(SM); float* Md = alloc(SM); float* My = alloc(SM); float* Ms = alloc(SM);
  float* X0 = alloc((size_t)Bn * 56 * 8192);
  float* hb1 = alloc(24576); float* hb2 = alloc(24576); float* hb3 = alloc(24576);
  float* gp = alloc(2048); float* f1 = alloc(1024); float* f2 = alloc(512);

  auto gemm = [&](const float* A, const float* Bm, float* C, const float* bias,
                  const float* Cres, int M, int N, int K,
                  long sAm, long sAk, long bA, long sBk, long sBn, long bB,
                  long sCm, long sCn, long bC, int batches, int act) {
    dim3 grid((unsigned)((M + 15) / 16), (unsigned)((N + 63) / 64), (unsigned)batches);
    k_gemm_wmma<<<grid, 128, 0, stream>>>(A, Bm, C, bias, Cres, M, N, K,
                                          sAm, sAk, bA, sBk, sBn, bB, sCm, sCn, bC, act);
  };
  auto g1 = [](long n) { return dim3((unsigned)((n + 255) / 256)); };

  // ---- stem: (2,1,32,32,64) -> (2,48,16,16,32)
  k_conv3d<<<g1((long)Bn * 48 * 16 * 16 * 32), 256, 0, stream>>>(
      P(IN_XA4C), P(ENC_STEM_W), P(ENC_STEM_B), xA, Bn, 1, 32, 32, 64, 48, 7, 2, 3, 16, 16, 32);
  float* x = xA; float* xAlt = xB;

  for (int i = 0; i < 4; ++i) {
    int C = FEAT[i]; long L = (long)DD[i] * HH[i] * WW[i];
    if (i > 0) {
      int Cp = FEAT[i - 1]; long Lp = (long)DD[i - 1] * HH[i - 1] * WW[i - 1];
      k_instnorm<<<Bn * Cp, 256, 0, stream>>>(x, t1, nullptr, Lp, 0);
      k_conv3d<<<g1((long)Bn * C * L), 256, 0, stream>>>(
          t1, P(ENC_DOWN_BASE + (i - 1) * 2 + 1), P(ENC_DOWN_BASE + (i - 1) * 2 + 0), xAlt,
          Bn, Cp, DD[i - 1], HH[i - 1], WW[i - 1], C, 2, 2, 0, DD[i], HH[i], WW[i]);
      float* tp = x; x = xAlt; xAlt = tp;
    }
    // ---- GSC block
    int gb = ENC_GSC_BASE + i * 8;
    long nCL = (long)Bn * C * L;
    k_conv3d<<<g1(nCL), 256, 0, stream>>>(x, P(gb + 4), P(gb + 0), t1,
        Bn, C, DD[i], HH[i], WW[i], C, 3, 1, 1, DD[i], HH[i], WW[i]);
    k_instnorm<<<Bn * C, 256, 0, stream>>>(t1, t1, nullptr, L, 1);
    k_conv3d<<<g1(nCL), 256, 0, stream>>>(t1, P(gb + 5), P(gb + 1), t2,
        Bn, C, DD[i], HH[i], WW[i], C, 3, 1, 1, DD[i], HH[i], WW[i]);
    k_instnorm<<<Bn * C, 256, 0, stream>>>(t2, t2, nullptr, L, 1);          // x1
    gemm(P(gb + 6), x, t1, P(gb + 2), nullptr, C, (int)L, C,
         C, 1, 0, L, 1, (long)C * L, L, 1, (long)C * L, Bn, 0);             // 1x1 conv w3
    k_instnorm<<<Bn * C, 256, 0, stream>>>(t1, t1, nullptr, L, 1);          // x2
    k_add<<<g1(nCL), 256, 0, stream>>>(t2, t1, t3, nCL);
    gemm(P(gb + 7), t3, t1, P(gb + 3), nullptr, C, (int)L, C,
         C, 1, 0, L, 1, (long)C * L, L, 1, (long)C * L, Bn, 0);             // 1x1 conv w4
    k_instnorm<<<Bn * C, 256, 0, stream>>>(t1, xAlt, x, L, 1);              // relu(IN)+x
    { float* tp = x; x = xAlt; xAlt = tp; }

    // ---- Mamba layers
    int dI = 2 * C, dtr = (C + 15) / 16, R = dtr + 32, ns = NS[i];
    long rowsZ = (long)Bn * 2 * dI, rowsM = (long)Bn * dI;
    for (int j = 0; j < 2; ++j) {
      int pb = ENC_STAGE_BASE + (i * 2 + j) * 25;
      long T = (long)Bn * L;
      k_layernorm<<<(unsigned)((T + 3) / 4), 128, 0, stream>>>(x, P(pb + 1), P(pb + 0), tok, C, L, T);
      gemm(P(pb + 16), tok, Z0, nullptr, nullptr, 2 * dI, (int)L, C,
           C, 1, 0, 1, C, (long)L * C, L, 1, (long)2 * dI * L, Bn, 0);      // in_proj
      for (int dir = 0; dir < 3; ++dir) {
        int db = (dir == 0) ? pb + 9 : (dir == 1 ? pb + 2 : pb + 18);       // fwd, bwd, sl
        const float* xzd = Z0;
        if (dir != 0) {
          k_gather<<<g1(rowsZ * L), 256, 0, stream>>>(Z0, Z1, rowsZ, L, dir == 1 ? 1 : 2, ns);
          xzd = Z1;
        }
        k_dwconv_silu<<<g1(rowsM * L), 256, 0, stream>>>(xzd, P(db + 3), P(db + 2), Mu, Bn, dI, L);
        gemm(P(db + 6), Mu, X0, nullptr, nullptr, R, (int)L, dI,
             dI, 1, 0, L, 1, (long)dI * L, L, 1, (long)R * L, Bn, 0);       // xproj
        k_dtproj_softplus<<<g1(rowsM * L), 256, 0, stream>>>(X0, P(db + 5), P(db + 4), Md,
                                                             Bn, dI, R, dtr, L);
        k_selscan<<<(unsigned)(Bn * (dI / 2)), 32, 0, stream>>>(Mu, Md, X0, xzd,
            P(db + 0), P(db + 1), My, Bn, dI, R, dtr, L);
        k_accum<<<g1(rowsM * L), 256, 0, stream>>>(My, Ms, rowsM, L, dir, ns);
      }
      gemm(P(pb + 17), Ms, xAlt, nullptr, x, C, (int)L, dI,
           dI, 1, 0, L, 1, (long)dI * L, L, 1, (long)C * L, Bn, 0);         // out_proj + residual
      { float* tp = x; x = xAlt; xAlt = tp; }
    }
  }

  // ---- final norm + MLP3 (1x1 convs over 16 voxels)
  k_instnorm<<<Bn * 384, 256, 0, stream>>>(x, t1, nullptr, 16, 0);
  gemm(P(ENC_MLP3_W1), t1, t2, P(ENC_MLP3_B1), nullptr, 768, 16, 384,
       384, 1, 0, 16, 1, (long)384 * 16, 16, 1, (long)768 * 16, Bn, 0);
  k_gelu<<<g1((long)Bn * 768 * 16), 256, 0, stream>>>(t2, (long)Bn * 768 * 16);
  gemm(P(ENC_MLP3_W2), t2, t1, P(ENC_MLP3_B2), nullptr, 384, 16, 768,
       768, 1, 0, 16, 1, (long)768 * 16, 16, 1, (long)384 * 16, Bn, 0);

  // ---- head
  gemm(P(HEAD_RES_W), t1, hb1, P(HEAD_RES_B), nullptr, 768, 16, 384,
       384, 1, 0, 16, 1, (long)384 * 16, 16, 1, (long)768 * 16, Bn, 0);
  k_instnorm<<<Bn * 768, 256, 0, stream>>>(hb1, hb1, nullptr, 16, 0);       // r
  k_conv3d<<<g1((long)Bn * 768 * 16), 256, 0, stream>>>(t1, P(HEAD_C1_W), P(HEAD_C1_B), hb2,
      Bn, 384, 2, 2, 4, 768, 3, 1, 1, 2, 2, 4);
  k_instnorm<<<Bn * 768, 256, 0, stream>>>(hb2, hb2, nullptr, 16, 2);       // lrelu
  k_conv3d<<<g1((long)Bn * 768 * 16), 256, 0, stream>>>(hb2, P(HEAD_C2_W), P(HEAD_C2_B), hb3,
      Bn, 768, 2, 2, 4, 768, 3, 1, 1, 2, 2, 4);
  k_instnorm<<<Bn * 768, 256, 0, stream>>>(hb3, hb3, nullptr, 16, 0);
  k_add_lrelu<<<g1((long)Bn * 768 * 16), 256, 0, stream>>>(hb3, hb1, hb2, (long)Bn * 768 * 16);

  // ---- classifier
  k_gap<<<g1((long)Bn * 768), 256, 0, stream>>>(hb2, gp, 16, (long)Bn * 768);
  gemm(P(CLS_FC1_W), gp, f1, P(CLS_FC1_B), nullptr, 512, 2, 768,
       768, 1, 0, 1, 768, 0, 1, 512, 0, 1, 0);
  k_bnrelu<<<g1(512), 256, 0, stream>>>(f1, P(CLS_BN1_G), P(CLS_BN1_B), 512, 1);
  gemm(P(CLS_FC2_W), f1, f2, P(CLS_FC2_B), nullptr, 256, 2, 512,
       512, 1, 0, 1, 512, 0, 1, 256, 0, 1, 0);
  k_bnrelu<<<g1(256), 256, 0, stream>>>(f2, P(CLS_BN2_G), P(CLS_BN2_B), 256, 1);
  gemm(P(CLS_FC3_W), f2, (float*)d_out, P(CLS_FC3_B), nullptr, 1, 2, 256,
       256, 1, 0, 1, 256, 0, 1, 1, 0, 1, 0);
}
